// MambaBlock_6854767805183
// MI455X (gfx1250) — compile-verified, matrix operand order checked by reading, not run on hardware
//
#include <hip/hip_runtime.h>
#include <hip/hip_bf16.h>
#include <math.h>

// ---------------------------------------------------------------------------
// Bidirectional Mamba block for MI455X (gfx1250, wave32, WMMA + async LDS).
// B=4, L=2048, D=256, DIN=512, DSTATE=16, DTR=16, DCONV=4.
// GEMMs: v_wmma_f32_16x16x32_bf16, weights staged to LDS via
// global_load_async_to_lds_b128 (ASYNCcnt) when the toolchain exposes it.
// ---------------------------------------------------------------------------

#define BB     4
#define LL     2048
#define DD     256
#define DIN    512
#define NPROJ  1024      // 2*DIN
#define DSTATE 16
#define DTR    16
#define DCONV  4
#define MROWS  (BB*LL)   // 8192
#define XDBLC  48        // DTR + 2*DSTATE

typedef __bf16 bf16_t;
typedef __attribute__((ext_vector_type(16))) __bf16 v16bf;
typedef __attribute__((ext_vector_type(8)))  __bf16 v8bf;
typedef __attribute__((ext_vector_type(8)))  float  v8f;

#if __has_builtin(__builtin_amdgcn_global_load_async_to_lds_b128)
#define HAVE_ASYNC_LDS 1
// Builtin signature (from hipcc diagnostic): (v4i AS1*, v4i AS3*, Ii, Ii)
typedef int v4i_t __attribute__((vector_size(16)));
typedef __attribute__((address_space(1))) v4i_t* gv4i_p;
typedef __attribute__((address_space(3))) v4i_t* lv4i_p;
#endif

// Stage `chunks` 16-byte chunks from global `src` to LDS `dst` (whole block).
__device__ __forceinline__ void stage_lds(const bf16_t* __restrict__ src,
                                          bf16_t* dst, int chunks) {
#ifdef HAVE_ASYNC_LDS
    for (int i = threadIdx.x; i < chunks; i += 256) {
        __builtin_amdgcn_global_load_async_to_lds_b128(
            (gv4i_p)((char*)(bf16_t*)src + (size_t)i * 16),
            (lv4i_p)((char*)dst + (size_t)i * 16), 0, 0);
    }
#if __has_builtin(__builtin_amdgcn_s_wait_asynccnt)
    __builtin_amdgcn_s_wait_asynccnt(0);
#else
    asm volatile("s_wait_asynccnt 0" ::: "memory");
#endif
#else
    for (int i = threadIdx.x; i < chunks; i += 256)
        ((v8bf*)dst)[i] = ((const v8bf*)src)[i];
#endif
}

// Core: one wave accumulates a 16x(16*NSUB) strip. A from global (row-major,
// stride K), B subtiles from LDS tile (NSUB*16 rows x K, row-major).
// Lane layout per CDNA5 ISA §7.12.2.
template<int NSUB>
__device__ __forceinline__ void gemm_core(const bf16_t* __restrict__ arow,
                                          const bf16_t* __restrict__ lds,
                                          int K, int r, int kb, v8f* acc) {
    #pragma unroll 2
    for (int k0 = 0; k0 < K; k0 += 32) {
        union { v16bf v; v8bf h[2]; } a;
        a.h[0] = *(const v8bf*)(arow + k0 + kb);
        a.h[1] = *(const v8bf*)(arow + k0 + 16 + kb);
        #pragma unroll
        for (int s = 0; s < NSUB; ++s) {
            union { v16bf v; v8bf h[2]; } b;
            const bf16_t* brow = lds + (size_t)(s * 16 + r) * K;
            b.h[0] = *(const v8bf*)(brow + k0 + kb);
            b.h[1] = *(const v8bf*)(brow + k0 + 16 + kb);
            acc[s] = __builtin_amdgcn_wmma_f32_16x16x32_bf16(
                         false, a.v, false, b.v, (short)0, acc[s], false, false);
        }
    }
}

// --------------------------- kernels ---------------------------------------

// RMSNorm row -> bf16.  One 64-thread block per row (2 waves); rev flips L.
__global__ void k_rmsnorm_bf16(const float* __restrict__ x,
                               const float* __restrict__ w,
                               bf16_t* __restrict__ xn, int rev) {
    const int row = blockIdx.x;                 // pipeline row: b*L + l
    const int b = row >> 11, lp = row & (LL - 1);
    const int ls = rev ? (LL - 1 - lp) : lp;
    const float* xr = x + ((size_t)(b * LL + ls)) * DD;
    const int t = threadIdx.x;                  // 0..63, 4 elems each
    float4 v = *(const float4*)(xr + t * 4);
    float ss = v.x * v.x + v.y * v.y + v.z * v.z + v.w * v.w;
    #pragma unroll
    for (int o = 16; o > 0; o >>= 1) ss += __shfl_down(ss, o, 32);
    __shared__ float sred[2];
    if ((t & 31) == 0) sred[t >> 5] = ss;
    __syncthreads();
    const float tot = sred[0] + sred[1];
    const float scale = rsqrtf(tot * (1.0f / (float)DD) + 1e-5f);
    float4 wv = *(const float4*)(w + t * 4);
    bf16_t* o4 = xn + (size_t)row * DD + t * 4;
    o4[0] = (bf16_t)(v.x * scale * wv.x);
    o4[1] = (bf16_t)(v.y * scale * wv.y);
    o4[2] = (bf16_t)(v.z * scale * wv.z);
    o4[3] = (bf16_t)(v.w * scale * wv.w);
}

// f32 -> bf16 weight conversion.
__global__ void k_cvt_bf16(const float* __restrict__ in,
                           bf16_t* __restrict__ out, int n) {
    int i = blockIdx.x * blockDim.x + threadIdx.x;
    if (i < n) out[i] = (bf16_t)in[i];
}

// Generic GEMM: C[M,N] = A[M,K] * W[N,K]^T.
// Block = 8 waves; block tile = 128(M) x 16*NSUB(N); W tile staged in LDS.
template<int NSUB>
__global__ void __launch_bounds__(256)
k_gemm_lds(const bf16_t* __restrict__ A, const bf16_t* __restrict__ W,
           float* __restrict__ C, int M, int N, int K) {
    extern __shared__ bf16_t smem[];            // NSUB*16*K bf16
    const int mgroups = M >> 7;                 // M / 128
    const int mg = blockIdx.x % mgroups;
    const int ng = blockIdx.x / mgroups;

    // Stage NSUB*16 consecutive weight rows (contiguous block) into LDS.
    const bf16_t* wt = W + (size_t)ng * (NSUB * 16) * K;
    stage_lds(wt, smem, (NSUB * 16 * K) / 8);
    __syncthreads();

    const int lane = threadIdx.x & 31;
    const int wv   = threadIdx.x >> 5;          // 0..7
    const int m0   = (mg * 8 + wv) * 16;
    const int r    = lane & 15;
    const int kb   = (lane >> 4) << 3;          // 0 or 8

    v8f acc[NSUB] = {};
    gemm_core<NSUB>(A + (size_t)(m0 + r) * K, smem, K, r, kb, acc);

    const int rbase = m0 + ((lane >> 4) << 3);
    #pragma unroll
    for (int s = 0; s < NSUB; ++s) {
        const int col = ng * (NSUB * 16) + s * 16 + (lane & 15);
        #pragma unroll
        for (int rr = 0; rr < 8; ++rr)
            C[(size_t)(rbase + rr) * N + col] = acc[s][rr];
    }
}

// outproj GEMM (M=8192, N=256, K=512) + residual + concat store into d_out.
__global__ void __launch_bounds__(256)
k_gemm_out_lds(const bf16_t* __restrict__ Y, const bf16_t* __restrict__ W,
               const float* __restrict__ x, float* __restrict__ out,
               int rev, int dirOff) {
    extern __shared__ bf16_t smem[];            // 64 * 512 bf16
    const int mgroups = MROWS >> 7;             // 64
    const int mg = blockIdx.x % mgroups;
    const int ng = blockIdx.x / mgroups;        // 0..3

    stage_lds(W + (size_t)ng * 64 * DIN, smem, (64 * DIN) / 8);
    __syncthreads();

    const int lane = threadIdx.x & 31;
    const int wv   = threadIdx.x >> 5;
    const int m0   = (mg * 8 + wv) * 16;
    const int r    = lane & 15;
    const int kb   = (lane >> 4) << 3;

    v8f acc[4] = {};
    gemm_core<4>(Y + (size_t)(m0 + r) * DIN, smem, DIN, r, kb, acc);

    const int rbase = m0 + ((lane >> 4) << 3);
    #pragma unroll
    for (int s = 0; s < 4; ++s) {
        const int col = ng * 64 + s * 16 + (lane & 15);
        #pragma unroll
        for (int rr = 0; rr < 8; ++rr) {
            const int m  = rbase + rr;
            const int b  = m >> 11;
            const int lp = m & (LL - 1);
            const int lo = rev ? (LL - 1 - lp) : lp;     // undo reversal
            const size_t rowo = (size_t)(b * LL + lo);
            out[rowo * (2 * DD) + dirOff + col] = acc[s][rr] + x[rowo * DD + col];
        }
    }
}

// Depthwise causal conv (DCONV=4) + bias + silu -> u (f32 + bf16 copies).
__global__ void k_conv_silu(const float* __restrict__ xh,
                            const float* __restrict__ cw,
                            const float* __restrict__ cb,
                            float* __restrict__ u,
                            bf16_t* __restrict__ ub) {
    const int idx = blockIdx.x * blockDim.x + threadIdx.x;   // B*L*DIN
    if (idx >= BB * LL * DIN) return;
    const int c = idx & (DIN - 1);
    const int l = (idx >> 9) & (LL - 1);
    const int b = idx >> 20;
    float acc = cb[c];
    const float* w4 = cw + c * DCONV;
    #pragma unroll
    for (int t = 0; t < DCONV; ++t) {
        const int ls = l - (DCONV - 1) + t;
        if (ls >= 0)
            acc = fmaf(xh[((size_t)(b * LL + ls)) * DIN + c], w4[t], acc);
    }
    const float s = acc / (1.0f + __expf(-acc));             // silu
    u[idx]  = s;
    ub[idx] = (bf16_t)s;
}

// dtproj (K=16, fp32 VALU) + bias + softplus.  m derived from blockIdx only
// so the x_dbl row pointer is provably uniform (SMEM path).
__global__ void k_dtproj_softplus(const float* __restrict__ xdbl,
                                  const float* __restrict__ w,
                                  const float* __restrict__ bias,
                                  float* __restrict__ dt) {
    const int m = blockIdx.x >> 1;                           // 0..8191
    const int n = ((blockIdx.x & 1) << 8) + threadIdx.x;     // 0..511
    const float* a  = xdbl + (size_t)m * XDBLC;              // uniform
    const float* wr = w + n * DTR;
    float s = bias[n];
    #pragma unroll
    for (int k = 0; k < DTR; ++k) s = fmaf(a[k], wr[k], s);
    dt[(size_t)m * DIN + n] = (s > 20.0f) ? s : log1pf(__expf(s));
}

// Selective scan fused with skip (u*D) and gate (silu(z)); writes y as bf16.
// One lane per (b, channel); 16 states in registers; b uniform per block so
// B/C row loads are wave-uniform (scalar-load friendly).
__global__ void k_scan_fuse(const float* __restrict__ dt,
                            const float* __restrict__ u,
                            const float* __restrict__ z,
                            const float* __restrict__ xdbl,
                            const float* __restrict__ A_log,
                            const float* __restrict__ Dp,
                            bf16_t* __restrict__ ybf) {
    const int b = blockIdx.x >> 1;                           // uniform
    const int d = ((blockIdx.x & 1) << 8) + threadIdx.x;     // 0..511
    float A[DSTATE], h[DSTATE];
    #pragma unroll
    for (int n = 0; n < DSTATE; ++n) {
        A[n] = -__expf(A_log[d * DSTATE + n]);
        h[n] = 0.0f;
    }
    const float Dd = Dp[d];
    const float* bc = xdbl + (size_t)(b * LL) * XDBLC;       // uniform
    for (int l = 0; l < LL; ++l, bc += XDBLC) {
        const size_t off = ((size_t)(b * LL + l)) * DIN + d;
        const float dtv = dt[off];
        const float uv  = u[off];
        const float zv  = z[off];
        __builtin_prefetch(bc + XDBLC, 0, 0);                // next step B/C
        const float dub = dtv * uv;
        float y = 0.0f;
        #pragma unroll
        for (int n = 0; n < DSTATE; ++n) {
            const float dA = __expf(dtv * A[n]);
            h[n] = fmaf(dA, h[n], dub * bc[DTR + n]);
            y    = fmaf(h[n], bc[DTR + DSTATE + n], y);
        }
        float yv = fmaf(uv, Dd, y);
        yv *= zv / (1.0f + __expf(-zv));                     // * silu(z)
        ybf[off] = (bf16_t)yv;
    }
}

// --------------------------- host side -------------------------------------

extern "C" void kernel_launch(void* const* d_in, const int* in_sizes, int n_in,
                              void* d_out, int out_size, void* d_ws, size_t ws_size,
                              hipStream_t stream) {
    (void)in_sizes; (void)n_in; (void)out_size; (void)ws_size;
    const float* x = (const float*)d_in[0];
    float* out = (float*)d_out;

    char* ws = (char*)d_ws;
    size_t off = 0;
    auto alloc = [&](size_t bytes) -> char* {
        off = (off + 255) & ~(size_t)255;
        char* p = ws + off;
        off += bytes;
        return p;
    };
    bf16_t* xn    = (bf16_t*)alloc((size_t)MROWS * DD * 2);        // 4 MB
    bf16_t* w_in  = (bf16_t*)alloc((size_t)NPROJ * DD * 2);        // 0.5 MB
    bf16_t* w_xp  = (bf16_t*)alloc((size_t)XDBLC * DIN * 2);       // 48 KB
    bf16_t* w_out = (bf16_t*)alloc((size_t)DD * DIN * 2);          // 256 KB
    float*  buf1  = (float*)alloc((size_t)MROWS * DIN * 4);        // xh -> dt
    float*  buf2  = (float*)alloc((size_t)MROWS * DIN * 4);        // z
    float*  buf3  = (float*)alloc((size_t)MROWS * DIN * 4);        // u (f32)
    bf16_t* buf4  = (bf16_t*)alloc((size_t)MROWS * DIN * 2);       // u_bf16 -> y_bf16
    float*  xdbl  = (float*)alloc((size_t)MROWS * XDBLC * 4);      // 1.5 MB

    for (int dir = 0; dir < 2; ++dir) {
        const float* norm_w   = (const float*)d_in[1  + dir * 10];
        const float* inproj_w = (const float*)d_in[2  + dir * 10];
        const float* conv_w   = (const float*)d_in[3  + dir * 10];
        const float* conv_b   = (const float*)d_in[4  + dir * 10];
        const float* xproj_w  = (const float*)d_in[5  + dir * 10];
        const float* dtproj_w = (const float*)d_in[6  + dir * 10];
        const float* dtproj_b = (const float*)d_in[7  + dir * 10];
        const float* A_log    = (const float*)d_in[8  + dir * 10];
        const float* Dp       = (const float*)d_in[9  + dir * 10];
        const float* outp_w   = (const float*)d_in[10 + dir * 10];
        const int rev = dir;

        // 1) rmsnorm (optionally reversed) -> bf16
        k_rmsnorm_bf16<<<MROWS, 64, 0, stream>>>(x, norm_w, xn, rev);

        // 2) weight conversions
        k_cvt_bf16<<<(NPROJ * DD + 255) / 256, 256, 0, stream>>>(inproj_w, w_in, NPROJ * DD);
        k_cvt_bf16<<<(XDBLC * DIN + 255) / 256, 256, 0, stream>>>(xproj_w, w_xp, XDBLC * DIN);
        k_cvt_bf16<<<(DD * DIN + 255) / 256, 256, 0, stream>>>(outp_w, w_out, DD * DIN);

        // 3) in_proj (WMMA+LDS): xh = xn*Win[0:512]^T ; z = xn*Win[512:1024]^T
        {
            const int grid = (MROWS / 128) * (DIN / 64);     // 64 * 8 = 512
            const size_t shmem = (size_t)64 * DD * 2;        // 32 KB
            k_gemm_lds<4><<<grid, 256, shmem, stream>>>(xn, w_in, buf1,
                                                        MROWS, DIN, DD);
            k_gemm_lds<4><<<grid, 256, shmem, stream>>>(xn, w_in + (size_t)DIN * DD,
                                                        buf2, MROWS, DIN, DD);
        }

        // 4) depthwise conv + bias + silu -> u (f32 + bf16)
        k_conv_silu<<<(MROWS * DIN + 255) / 256, 256, 0, stream>>>(buf1, conv_w, conv_b,
                                                                   buf3, buf4);

        // 5) xproj (WMMA+LDS): x_dbl = u * Wxp^T   (N=48, one n-group)
        {
            const int grid = (MROWS / 128) * 1;              // 64
            const size_t shmem = (size_t)48 * DIN * 2;       // 48 KB
            k_gemm_lds<3><<<grid, 256, shmem, stream>>>(buf4, w_xp, xdbl,
                                                        MROWS, XDBLC, DIN);
        }

        // 6) dtproj + softplus (fp32, uniform row pointer) -> dt (reuses buf1)
        k_dtproj_softplus<<<MROWS * 2, 256, 0, stream>>>(xdbl, dtproj_w,
                                                         dtproj_b, buf1);

        // 7) selective scan fused with skip + gate -> y bf16 (reuses buf4)
        k_scan_fuse<<<BB * 2, 256, 0, stream>>>(buf1, buf3, buf2, xdbl,
                                                A_log, Dp, buf4);

        // 8) outproj (WMMA+LDS) + residual + concat store
        {
            const int grid = (MROWS / 128) * (DD / 64);      // 64 * 4 = 256
            const size_t shmem = (size_t)64 * DIN * 2;       // 64 KB
            k_gemm_out_lds<<<grid, 256, shmem, stream>>>(buf4, w_out, x, out,
                                                         rev, dir * DD);
        }
    }
}
